// CA_SH_ProteinFeatures_3607772528735
// MI455X (gfx1250) — matrix operand order repaired and checked by reading, not compiled, not run on hardware
//
#include <hip/hip_runtime.h>
#include <math.h>

typedef _Float16 half_t;
typedef __attribute__((ext_vector_type(16))) _Float16 v16h;
typedef __attribute__((ext_vector_type(8)))  _Float16 v8h;
typedef __attribute__((ext_vector_type(8)))  float    v8f;

#define KNB   30
#define CIN   163
#define CPAD  192
#define COUT  128
#define NRBF  16
#define MAXREL 32

// ---------------------------------------------------------------------------
// Kernel 0: convert W_edge [128,163] f32 -> f16 padded [128,192] (row-major)
// ---------------------------------------------------------------------------
__global__ void wconv_kernel(const float* __restrict__ W, half_t* __restrict__ Wf) {
    int t = blockIdx.x * blockDim.x + threadIdx.x;
    if (t >= COUT * CPAD) return;
    int n = t / CPAD, k = t - n * CPAD;
    Wf[t] = (k < CIN) ? (half_t)W[n * CIN + k] : (half_t)0.0f;
}

// ---------------------------------------------------------------------------
// Kernel 1: spherical harmonic l2-norms per node (closed form)
// ---------------------------------------------------------------------------
__global__ void sh_kernel(const float* __restrict__ Ca, float* __restrict__ SH, int n) {
    int t = blockIdx.x * blockDim.x + threadIdx.x;
    if (t >= n) return;
    float x = Ca[3*t], y = Ca[3*t+1], z = Ca[3*t+2];
    float r = sqrtf(x*x + y*y + z*z);
    float phi = atan2f(y, x);
    float cp = cosf(phi), cp2 = cp*cp, s2 = 1.0f - cp2;
    const float I4P = 0.07957747154594767f;            // 1/(4*pi)
    // l = 0
    float f0 = I4P;
    // l = 1 : m=-1 A=3/(4pi)*2, extra (1/2)^2 on P^2 ; m=0 A=3/(4pi) ; m=1 A=3/(8pi)
    float A1m = 3.0f*I4P*2.0f, A10 = 3.0f*I4P, A1p = 3.0f*I4P*0.5f;
    float acc1 = A1m*A1m * cp2 * (s2 * 0.25f)
               + A10*A10 * cp2
               + A1p*A1p * cp2 * s2;
    float f1 = sqrtf(acc1);
    // l = 2
    float A2m2 = 5.0f*I4P*24.0f, A2m1 = 5.0f*I4P*6.0f, A20 = 5.0f*I4P;
    float A2p1 = 5.0f*I4P/6.0f,  A2p2 = 5.0f*I4P/24.0f;
    float c2 = cosf(2.0f*phi);  float c2sq = c2*c2;
    float p20 = 0.5f*(3.0f*cp2 - 1.0f); float p20sq = p20*p20;
    float p21 = 9.0f*cp2*s2;
    float p22 = 9.0f*s2*s2;
    float acc2 = A2m2*A2m2 * c2sq * p22 * (1.0f/576.0f)
               + A2m1*A2m1 * cp2  * p21 * (1.0f/36.0f)
               + A20*A20   * p20sq
               + A2p1*A2p1 * cp2  * p21
               + A2p2*A2p2 * c2sq * p22;
    float f2 = sqrtf(acc2);
    float zr = z / r;
    bool bad = !(fabsf(zr) <= 1.0f);                    // catches r==0 (NaN) too
    SH[3*t+0] = bad ? 0.0f : f0;
    SH[3*t+1] = (bad || !(f1 == f1)) ? 0.0f : f1;
    SH[3*t+2] = (bad || !(f2 == f2)) ? 0.0f : f2;
}

// ---------------------------------------------------------------------------
// Kernel 2: masked pairwise distances + iterative top-30 (one WG per (b,i))
// ---------------------------------------------------------------------------
__global__ __launch_bounds__(256) void topk_kernel(
    const float* __restrict__ Ca, const float* __restrict__ mask,
    float* __restrict__ outIdxF, float* __restrict__ Dn, int* __restrict__ IdxI, int L)
{
    __shared__ float sD[2048];
    __shared__ float rv[256];
    __shared__ int   ri[256];
    __shared__ int   selI[KNB];
    __shared__ float selV[KNB];
    int bi = blockIdx.x;
    int b = bi / L, i = bi - b * L;
    int tid = threadIdx.x;
    const float* cab = Ca + (size_t)b * L * 3;
    const float* mb  = mask + (size_t)b * L;
    float xi = cab[3*i], yi = cab[3*i+1], zi = cab[3*i+2];
    float mi = mb[i];

    float lmax = 0.0f;
    for (int j = tid; j < L; j += 256) {
        float dx = xi - cab[3*j], dy = yi - cab[3*j+1], dz = zi - cab[3*j+2];
        float m2 = mi * mb[j];
        float d = m2 * sqrtf(dx*dx + dy*dy + dz*dz + 1e-6f);
        sD[j] = d;
        lmax = fmaxf(lmax, d);
    }
    rv[tid] = lmax; __syncthreads();
    for (int s = 128; s > 0; s >>= 1) {
        if (tid < s) rv[tid] = fmaxf(rv[tid], rv[tid + s]);
        __syncthreads();
    }
    float dmax = rv[0];
    __syncthreads();
    for (int j = tid; j < L; j += 256) {
        float m2 = mi * mb[j];
        sD[j] = sD[j] + (1.0f - m2) * dmax;
    }
    __syncthreads();

    for (int k = 0; k < KNB; ++k) {
        float bv = 3.4e38f; int bj = 0x7fffffff;
        for (int j = tid; j < L; j += 256) {
            float v = sD[j];
            if (v < bv) { bv = v; bj = j; }
        }
        rv[tid] = bv; ri[tid] = bj; __syncthreads();
        for (int s = 128; s > 0; s >>= 1) {
            if (tid < s) {
                float v2 = rv[tid + s]; int j2 = ri[tid + s];
                if (v2 < rv[tid] || (v2 == rv[tid] && j2 < ri[tid])) { rv[tid] = v2; ri[tid] = j2; }
            }
            __syncthreads();
        }
        if (tid == 0) { selI[k] = ri[0]; selV[k] = rv[0]; sD[ri[0]] = 3.4e38f; }
        __syncthreads();
    }
    if (tid < KNB) {
        size_t o = (size_t)bi * KNB + tid;
        Dn[o] = selV[tid];
        IdxI[o] = selI[tid];
        outIdxF[o] = (float)selI[tid];
    }
}

// ---------------------------------------------------------------------------
// Kernel 3: edge features -> WMMA GEMM (163->128, padded 192) -> LayerNorm
// One workgroup (128 thr = 4 waves) per (b,i). wmma f32_16x16x32_f16.
// ---------------------------------------------------------------------------
__device__ __forceinline__ float pdist3(const float* a, const float* bp) {
    float dx = a[0]-bp[0], dy = a[1]-bp[1], dz = a[2]-bp[2];
    return sqrtf(dx*dx + dy*dy + dz*dz + 1e-6f);
}
__device__ __forceinline__ void put_rbf(half_t* row, int base, float D) {
    const float inv_sigma = 1.0f / 1.25f;               // sigma = 20/16
#pragma unroll
    for (int t = 0; t < NRBF; ++t) {
        float mu = 2.0f + (20.0f / 15.0f) * (float)t;   // linspace(2,22,16)
        float u = (D - mu) * inv_sigma;
        row[base + t] = (half_t)__expf(-u * u);
    }
}

__global__ __launch_bounds__(128) void edge_kernel(
    const float* __restrict__ Ca, const int* __restrict__ resid,
    const int* __restrict__ chains,
    const float* __restrict__ W_pos, const float* __restrict__ b_pos,
    const float* __restrict__ gamma, const float* __restrict__ beta,
    const float* __restrict__ Dn, const int* __restrict__ IdxI,
    const float* __restrict__ SH, const half_t* __restrict__ Wf,
    float* __restrict__ outE, int L)
{
    __shared__ __align__(16) half_t Af[32 * CPAD];      // A panel: 32 edges x 192 feats
    __shared__ float Cout[32 * COUT];
    __shared__ float s_mu[32], s_rs[32];
    int bi = blockIdx.x;
    int b = bi / L, i = bi - b * L;
    int tid = threadIdx.x;

    // zero-pad A panel (rows 30,31 and K 163..191 stay zero)
    for (int t = tid; t < 32 * CPAD / 2; t += 128) ((unsigned int*)Af)[t] = 0u;
    __syncthreads();

    if (tid < KNB) {
        int e = tid;
        int j = IdxI[(size_t)bi * KNB + e];
        const float* cab = Ca + (size_t)b * L * 3;
        float ap[3], ac[3], an[3], bp[3], bc[3], bn[3];
#pragma unroll
        for (int c = 0; c < 3; ++c) {
            ac[c] = cab[3*i + c];
            ap[c] = (i == 0)     ? 0.0f : cab[3*(i-1) + c];
            an[c] = (i == L-1)   ? 0.0f : cab[3*(i+1) + c];
            bc[c] = cab[3*j + c];
            bp[c] = (j == 0)     ? 0.0f : cab[3*(j-1) + c];
            bn[c] = (j == L-1)   ? 0.0f : cab[3*(j+1) + c];
        }
        half_t* row = Af + e * CPAD;
        // positional encoding: one-hot(66) @ W_pos.T + b_pos == column gather
        int off = resid[b*L + i] - resid[b*L + j];
        int same = (chains[b*L + i] == chains[b*L + j]) ? 1 : 0;
        int dcl = off + MAXREL; dcl = dcl < 0 ? 0 : (dcl > 2*MAXREL ? 2*MAXREL : dcl);
        int dsel = same ? dcl : (2*MAXREL + 1);
#pragma unroll
        for (int c = 0; c < 16; ++c)
            row[c] = (half_t)(W_pos[c * 66 + dsel] + b_pos[c]);
        // 9 RBF sets (order matches reference concat)
        put_rbf(row, 16,  Dn[(size_t)bi * KNB + e]);
        put_rbf(row, 32,  pdist3(ap, bp));
        put_rbf(row, 48,  pdist3(an, bn));
        put_rbf(row, 64,  pdist3(ap, bc));
        put_rbf(row, 80,  pdist3(ap, bn));
        put_rbf(row, 96,  pdist3(ac, bp));
        put_rbf(row, 112, pdist3(ac, bn));
        put_rbf(row, 128, pdist3(an, bp));
        put_rbf(row, 144, pdist3(an, bc));
        // SH of neighbor
        row[160] = (half_t)SH[3*((size_t)b*L + j) + 0];
        row[161] = (half_t)SH[3*((size_t)b*L + j) + 1];
        row[162] = (half_t)SH[3*((size_t)b*L + j) + 2];
    }
    __syncthreads();

    // ---- WMMA GEMM: [32 x 192] x [192 x 128] -> [32 x 128] ----
    int wave = tid >> 5, lane = tid & 31;
    int lr = lane & 15, hi = lane >> 4;
    v8f acc[2][2] = {};
#pragma unroll
    for (int ks = 0; ks < CPAD / 32; ++ks) {
        int kb = ks * 32;
        v16h afr[2];
#pragma unroll
        for (int mt = 0; mt < 2; ++mt) {
            const half_t* base = Af + (mt*16 + lr) * CPAD + kb;
            union { v16h v; v8h h[2]; } u;
            u.h[0] = *(const v8h*)(base + 8*hi);        // K = kb + 8*hi + 0..7
            u.h[1] = *(const v8h*)(base + 16 + 8*hi);   // K = kb + 16 + 8*hi + 0..7
            afr[mt] = u.v;
        }
#pragma unroll
        for (int nt = 0; nt < 2; ++nt) {
            int n = (wave * 2 + nt) * 16 + lr;
            union { v16h v; v8h h[2]; } ub;
            ub.h[0] = *(const v8h*)(Wf + (size_t)n * CPAD + kb + 8*hi);
            ub.h[1] = *(const v8h*)(Wf + (size_t)n * CPAD + kb + 16 + 8*hi);
            v16h bfr = ub.v;
            acc[0][nt] = __builtin_amdgcn_wmma_f32_16x16x32_f16(
                false, afr[0], false, bfr, (short)0, acc[0][nt], false, false);
            acc[1][nt] = __builtin_amdgcn_wmma_f32_16x16x32_f16(
                false, afr[1], false, bfr, (short)0, acc[1][nt], false, false);
        }
    }
    // spill accumulators: VGPR r -> row (r + 8*hi), col = lane & 15
#pragma unroll
    for (int mt = 0; mt < 2; ++mt)
#pragma unroll
        for (int nt = 0; nt < 2; ++nt) {
            int col = (wave * 2 + nt) * 16 + lr;
#pragma unroll
            for (int r = 0; r < 8; ++r)
                Cout[(mt*16 + r + 8*hi) * COUT + col] = acc[mt][nt][r];
        }
    __syncthreads();

    // ---- LayerNorm over 128 channels per edge ----
    if (tid < KNB) {
        float s = 0.0f, s2 = 0.0f;
#pragma unroll 4
        for (int c = 0; c < COUT; ++c) {
            float v = Cout[tid * COUT + c];
            s += v; s2 += v * v;
        }
        float mu = s * (1.0f / COUT);
        float var = s2 * (1.0f / COUT) - mu * mu;
        s_mu[tid] = mu;
        s_rs[tid] = rsqrtf(var + 1e-5f);
    }
    __syncthreads();
    for (int t = tid; t < KNB * COUT; t += 128) {
        int e = t >> 7, c = t & 127;
        float v = (Cout[e * COUT + c] - s_mu[e]) * s_rs[e] * gamma[c] + beta[c];
        outE[((size_t)bi * KNB + e) * COUT + c] = v;
    }
}

// ---------------------------------------------------------------------------
extern "C" void kernel_launch(void* const* d_in, const int* in_sizes, int n_in,
                              void* d_out, int out_size, void* d_ws, size_t ws_size,
                              hipStream_t stream) {
    const float* Ca     = (const float*)d_in[0];
    const float* mask   = (const float*)d_in[1];
    const int*   resid  = (const int*)d_in[2];
    const int*   chains = (const int*)d_in[3];
    const float* W_pos  = (const float*)d_in[4];
    const float* b_pos  = (const float*)d_in[5];
    const float* W_edge = (const float*)d_in[6];
    const float* gamma  = (const float*)d_in[7];
    const float* beta   = (const float*)d_in[8];

    int n_nodes = in_sizes[1];              // B*L
    int L = 2048;
    size_t nBLK = (size_t)n_nodes * KNB;

    // workspace layout
    float*  Dn   = (float*)d_ws;
    int*    IdxI = (int*)((char*)d_ws + nBLK * 4);
    float*  SH   = (float*)((char*)d_ws + nBLK * 8);
    half_t* Wf   = (half_t*)((char*)d_ws + nBLK * 8 + (size_t)n_nodes * 3 * 4);

    float* outE   = (float*)d_out;
    float* outIdx = outE + nBLK * COUT;

    wconv_kernel<<<(COUT * CPAD + 255) / 256, 256, 0, stream>>>(W_edge, Wf);
    sh_kernel<<<(n_nodes + 255) / 256, 256, 0, stream>>>(Ca, SH, n_nodes);
    topk_kernel<<<n_nodes, 256, 0, stream>>>(Ca, mask, outIdx, Dn, IdxI, L);
    edge_kernel<<<n_nodes, 128, 0, stream>>>(Ca, resid, chains, W_pos, b_pos,
                                             gamma, beta, Dn, IdxI, SH, Wf, outE, L);
}
